// SwinTransformerBlock_66236985639670
// MI455X (gfx1250) — matrix operand order repaired and checked by reading, not compiled, not run on hardware
//
#include <hip/hip_runtime.h>
#include <hip/hip_bf16.h>
#include <math.h>
#include <stdint.h>

typedef __bf16 bf16_t;
typedef __attribute__((ext_vector_type(16))) __bf16 v16bf;
typedef __attribute__((ext_vector_type(8)))  __bf16 v8bf;
typedef __attribute__((ext_vector_type(8)))  float  v8f;

#define WMMA_BF16(A_, B_, C_) \
  __builtin_amdgcn_wmma_f32_16x16x32_bf16(false, (A_), false, (B_), (short)0, (C_), false, false)

// CDNA5 async global->LDS copy path (ASYNCcnt-tracked), guarded for toolchain support.
#if defined(__has_builtin)
#if __has_builtin(__builtin_amdgcn_global_load_async_to_lds_b128) && \
    __has_builtin(__builtin_amdgcn_s_wait_asynccnt)
#define USE_ASYNC_LDS 1
#endif
#endif
#ifndef USE_ASYNC_LDS
#define USE_ASYNC_LDS 0
#endif

#if USE_ASYNC_LDS
typedef int v4i32 __attribute__((vector_size(16)));
typedef __attribute__((address_space(1))) v4i32 gv4i;  // prints as 'v4i32 __device__' in HIP
typedef __attribute__((address_space(3))) v4i32 lv4i;  // prints as 'v4i32 __shared__'
static __device__ __forceinline__ void async_copy_b128(const void* gsrc, void* ldst) {
  // flat->AS(3) is a low-32-bit truncation on AMDGCN, so the integer route matches addrspacecast
  __builtin_amdgcn_global_load_async_to_lds_b128(
      (gv4i*)(uintptr_t)gsrc, (lv4i*)(uint32_t)(uintptr_t)ldst, 0, 0);
}
#endif

// ---------------- problem constants ----------------
#define C_DIM   192
#define HEADS   6
#define CH      32
#define WIN     7
#define NTOK    49           // tokens per window
#define SHIFTV  3
#define IMG     224
#define BATCH   4
#define TOKENS  200704       // 4*224*224
#define NWTOT   4096         // total windows
#define QSCALE  0.17677669529663687f   // 32^-0.5

static __device__ __forceinline__ int region224(int u) {
  // slices (0,-7), (-7,-3), (-3,None) on a 224-long axis
  return (u < IMG - WIN) ? 0 : ((u < IMG - SHIFTV) ? 1 : 2);
}

static __device__ __forceinline__ float wave_reduce_sum(float v) {
#pragma unroll
  for (int off = 16; off > 0; off >>= 1) v += __shfl_xor(v, off, 32);
  return v;
}

// ---------------- weight convert/transpose: (K,N) fp32 -> (N,K) bf16 ----------------
__global__ void transpose_to_bf16(const float* __restrict__ w, bf16_t* __restrict__ wt,
                                  int K, int N) {
  int idx = blockIdx.x * 256 + threadIdx.x;
  if (idx < K * N) {
    int k = idx / N, n = idx % N;
    wt[(size_t)n * K + k] = (bf16_t)w[(size_t)k * N + n];
  }
}

// ---------------- LayerNorm (wave per row). WINDOWED: fused roll(-3,-3)+partition ----------------
template <bool WINDOWED>
__global__ __launch_bounds__(256) void layernorm_kernel(const float* __restrict__ x,
                                                        const float* __restrict__ g,
                                                        const float* __restrict__ bta,
                                                        bf16_t* __restrict__ out) {
  int row  = blockIdx.x * 8 + (threadIdx.x >> 5);
  int lane = threadIdx.x & 31;
  const float* src;
  if (WINDOWED) {
    int w = row / NTOK, i = row % NTOK;
    int b = w >> 10, wir = w & 1023;
    int wh = wir >> 5, ww = wir & 31;
    int r = i / WIN, c = i % WIN;
    int hp = (wh * WIN + r + SHIFTV) % IMG;
    int wp = (ww * WIN + c + SHIFTV) % IMG;
    src = x + ((size_t)b * (IMG * IMG) + hp * IMG + wp) * C_DIM;
  } else {
    src = x + (size_t)row * C_DIM;
  }
  float v[6];
  float s = 0.f;
#pragma unroll
  for (int j = 0; j < 6; ++j) { v[j] = src[lane + j * 32]; s += v[j]; }
  float mu = wave_reduce_sum(s) * (1.0f / C_DIM);
  float var = 0.f;
#pragma unroll
  for (int j = 0; j < 6; ++j) { float d = v[j] - mu; var += d * d; }
  var = wave_reduce_sum(var) * (1.0f / C_DIM);
  float rstd = rsqrtf(var + 1e-5f);
  bf16_t* dst = out + (size_t)row * C_DIM;
#pragma unroll
  for (int j = 0; j < 6; ++j) {
    int c = lane + j * 32;
    dst[c] = (bf16_t)((v[j] - mu) * rstd * g[c] + bta[c]);
  }
}

// ---------------- epilogues ----------------
struct QkvEpi {
  const float* bias; bf16_t* out;   // out: (TOKENS, 576)
  __device__ void operator()(int m, int n, float v) const {
    v += bias[n];
    if (n < C_DIM) v *= QSCALE;     // pre-scale q
    out[(size_t)m * 576 + n] = (bf16_t)v;
  }
};
struct ProjEpi {                    // window-reverse + roll(+3,+3) + residual -> x2 fp32
  const float* bias; const float* x; float* x2;
  __device__ void operator()(int m, int n, float v) const {
    v += bias[n];
    int w = m / NTOK, i = m % NTOK;
    int b = w >> 10, wir = w & 1023;
    int wh = wir >> 5, ww = wir & 31;
    int r = i / WIN, c = i % WIN;
    int hp = (wh * WIN + r + SHIFTV) % IMG;
    int wp = (ww * WIN + c + SHIFTV) % IMG;
    size_t idx = ((size_t)b * (IMG * IMG) + hp * IMG + wp) * C_DIM + n;
    x2[idx] = x[idx] + v;
  }
};
struct Fc1Epi {                     // bias + exact GELU -> bf16
  const float* bias; bf16_t* hid;
  __device__ void operator()(int m, int n, float v) const {
    v += bias[n];
    float g = 0.5f * v * (1.0f + erff(v * 0.70710678118654752f));
    hid[(size_t)m * 768 + n] = (bf16_t)g;
  }
};
struct Fc2Epi {                     // bias + residual -> fp32 out
  const float* bias; const float* x2; float* out;
  __device__ void operator()(int m, int n, float v) const {
    size_t idx = (size_t)m * C_DIM + n;
    out[idx] = x2[idx] + v + bias[n];
  }
};

// ---------------- tiled bf16 WMMA GEMM: C(M,N) = A(M,K) * Bt(N,K)^T ----------------
// block tile 128x64, K-step 32; 8 waves (4x2), each wave 32x32 (2x2 WMMA tiles).
// Double-buffered LDS; tiles staged with CDNA5 async global->LDS copies when available.
template <typename Epi>
__global__ __launch_bounds__(256) void gemm_bf16_kernel(const bf16_t* __restrict__ A,
                                                        const bf16_t* __restrict__ Bt,
                                                        int K, Epi epi) {
  __shared__ __align__(64) bf16_t As[2][128][32];
  __shared__ __align__(64) bf16_t Bs[2][64][32];
  const int tid  = threadIdx.x;
  const int lane = tid & 31;
  const int wave = tid >> 5;
  const int waveM = wave >> 1, waveN = wave & 1;
  const int bm = blockIdx.x * 128, bn = blockIdx.y * 64;
  const int fr = lane & 15, half = lane >> 4;
  const int arow = tid >> 1, ah = tid & 1;   // A stage: 128 rows x two 16-elem chunks
  const int brow = tid >> 2, bq = tid & 3;   // B stage: 64 rows x four 8-elem chunks

  v8f acc[2][2] = {{{0.f}}};
  const int nk = K >> 5;

#if USE_ASYNC_LDS
  {  // prologue: fill buffer 0 (3 async b128 per thread)
    const bf16_t* ga = A + (size_t)(bm + arow) * K + ah * 16;
    const bf16_t* gb = Bt + (size_t)(bn + brow) * K + bq * 8;
    async_copy_b128(ga,     &As[0][arow][ah * 16]);
    async_copy_b128(ga + 8, &As[0][arow][ah * 16 + 8]);
    async_copy_b128(gb,     &Bs[0][brow][bq * 8]);
  }
#endif

  for (int it = 0; it < nk; ++it) {
    const int k0  = it << 5;
    const int cur = it & 1;
#if USE_ASYNC_LDS
    const int nxt = cur ^ 1;
    if (it + 1 < nk) {  // prefetch next K-slab into the other buffer, then drain cur's 3
      const bf16_t* ga = A + (size_t)(bm + arow) * K + (k0 + 32) + ah * 16;
      const bf16_t* gb = Bt + (size_t)(bn + brow) * K + (k0 + 32) + bq * 8;
      async_copy_b128(ga,     &As[nxt][arow][ah * 16]);
      async_copy_b128(ga + 8, &As[nxt][arow][ah * 16 + 8]);
      async_copy_b128(gb,     &Bs[nxt][brow][bq * 8]);
      __builtin_amdgcn_s_wait_asynccnt(3);
    } else {
      __builtin_amdgcn_s_wait_asynccnt(0);
    }
    __syncthreads();   // cur buffer fully resident for every wave
#else
    __syncthreads();
    *(v16bf*)(&As[cur][arow][ah * 16]) =
        *(const v16bf*)(A + (size_t)(bm + arow) * K + k0 + ah * 16);
    *(v8bf*)(&Bs[cur][brow][bq * 8]) =
        *(const v8bf*)(Bt + (size_t)(bn + brow) * K + k0 + bq * 8);
    if (k0 + 32 < K) {
      __builtin_prefetch(A + (size_t)(bm + arow) * K + k0 + 32, 0, 1);
      __builtin_prefetch(Bt + (size_t)(bn + brow) * K + k0 + 32, 0, 1);
    }
    __syncthreads();
#endif

    v16bf afrag[2], bfrag[2];
#pragma unroll
    for (int tm = 0; tm < 2; ++tm) {       // A 16x32 lane map: K in {half*8..} U {16+half*8..}
      int r = waveM * 32 + tm * 16 + fr;
      v8bf lo = *(const v8bf*)(&As[cur][r][half * 8]);
      v8bf hi = *(const v8bf*)(&As[cur][r][16 + half * 8]);
      afrag[tm] = __builtin_shufflevector(lo, hi, 0,1,2,3,4,5,6,7,8,9,10,11,12,13,14,15);
    }
#pragma unroll
    for (int tn = 0; tn < 2; ++tn) {       // B 32x16 lane map: n=l%16, K in {half*8..} U {16+half*8..}
      int r = waveN * 32 + tn * 16 + fr;
      v8bf lo = *(const v8bf*)(&Bs[cur][r][half * 8]);
      v8bf hi = *(const v8bf*)(&Bs[cur][r][16 + half * 8]);
      bfrag[tn] = __builtin_shufflevector(lo, hi, 0,1,2,3,4,5,6,7,8,9,10,11,12,13,14,15);
    }
#pragma unroll
    for (int tm = 0; tm < 2; ++tm)
#pragma unroll
      for (int tn = 0; tn < 2; ++tn)
        acc[tm][tn] = WMMA_BF16(afrag[tm], bfrag[tn], acc[tm][tn]);

#if USE_ASYNC_LDS
    __syncthreads();   // all waves done reading cur before it is refilled next iter
#endif
  }
  // C layout: vgpr r -> M = half*8 + r ; N = l%16
#pragma unroll
  for (int tm = 0; tm < 2; ++tm)
#pragma unroll
    for (int tn = 0; tn < 2; ++tn)
#pragma unroll
      for (int r = 0; r < 8; ++r) {
        int m = bm + waveM * 32 + tm * 16 + half * 8 + r;
        int n = bn + waveN * 32 + tn * 16 + fr;
        epi(m, n, acc[tm][tn][r]);
      }
}

// ---------------- attention: one wave per (window, head) ----------------
#define AT_PER_WAVE 36864   // Qs 4K + Ks 4K + Vt 4K + Ss 16K + Ps 8K (bytes)
__global__ __launch_bounds__(128) void attn_kernel(const bf16_t* __restrict__ qkv,
                                                   const float* __restrict__ rpb,
                                                   bf16_t* __restrict__ attn_out) {
  extern __shared__ __align__(128) char attn_smem[];
  const int wave = threadIdx.x >> 5, lane = threadIdx.x & 31;
  const int wh = blockIdx.x * 4 + wave;
  const int w = wh / HEADS, h = wh % HEADS;
  char* base = attn_smem + wave * AT_PER_WAVE;
  bf16_t (*Qs)[32] = (bf16_t(*)[32])(base);
  bf16_t (*Ks)[32] = (bf16_t(*)[32])(base + 4096);
  bf16_t (*Vt)[64] = (bf16_t(*)[64])(base + 8192);
  float  (*Ss)[64] = (float (*)[64])(base + 12288);
  bf16_t (*Ps)[64] = (bf16_t(*)[64])(base + 28672);
  const int fr = lane & 15, half = lane >> 4;

  // stage Q,K rows (zero-pad 49->64); q was pre-scaled in QKV epilogue
  for (int t = lane; t < 256; t += 32) {
    int row = t >> 2, ch = t & 3;
    uint4 zq = {0u, 0u, 0u, 0u}, zk = {0u, 0u, 0u, 0u};
    if (row < NTOK) {
      const bf16_t* p = qkv + ((size_t)(w * NTOK + row)) * 576 + h * CH;
      zq = ((const uint4*)p)[ch];
      zk = ((const uint4*)(p + C_DIM))[ch];
    }
    *(uint4*)(&Qs[row][ch * 8]) = zq;
    *(uint4*)(&Ks[row][ch * 8]) = zk;
  }
  // stage V transposed: Vt[c][j] = v[j][c], zero-pad columns
  for (int t = lane; t < 2048; t += 32) {
    int c = t >> 6, j = t & 63;
    bf16_t val = (bf16_t)0.0f;
    if (j < NTOK) val = qkv[((size_t)(w * NTOK + j)) * 576 + 2 * C_DIM + h * CH + c];
    Vt[c][j] = val;
  }
  __syncthreads();

  // S = Q * K^T  (64x64, K-dim 32): 4x4 WMMA tiles
  {
    v16bf aq[4], bk[4];
#pragma unroll
    for (int tm = 0; tm < 4; ++tm) {
      int r = tm * 16 + fr;
      v8bf lo = *(const v8bf*)(&Qs[r][half * 8]);
      v8bf hi = *(const v8bf*)(&Qs[r][16 + half * 8]);
      aq[tm] = __builtin_shufflevector(lo, hi, 0,1,2,3,4,5,6,7,8,9,10,11,12,13,14,15);
    }
#pragma unroll
    for (int tn = 0; tn < 4; ++tn) {
      v8bf lo = *(const v8bf*)(&Ks[tn * 16 + fr][half * 8]);
      v8bf hi = *(const v8bf*)(&Ks[tn * 16 + fr][16 + half * 8]);
      bk[tn] = __builtin_shufflevector(lo, hi, 0,1,2,3,4,5,6,7,8,9,10,11,12,13,14,15);
    }
    v8f sacc[4][4] = {{{0.f}}};
#pragma unroll
    for (int tm = 0; tm < 4; ++tm)
#pragma unroll
      for (int tn = 0; tn < 4; ++tn)
        sacc[tm][tn] = WMMA_BF16(aq[tm], bk[tn], sacc[tm][tn]);
#pragma unroll
    for (int tm = 0; tm < 4; ++tm)
#pragma unroll
      for (int tn = 0; tn < 4; ++tn)
#pragma unroll
        for (int r = 0; r < 8; ++r)
          Ss[tm * 16 + half * 8 + r][tn * 16 + fr] = sacc[tm][tn][r];
  }
  __syncthreads();

  // softmax rows (each lane owns rows lane, lane+32), inline rel-pos bias + shift mask
  {
    int wir = w & 1023, whh = wir >> 5, www = wir & 31;
#pragma unroll
    for (int rep = 0; rep < 2; ++rep) {
      int m = lane + rep * 32;
      if (m >= NTOK) {
        for (int j = 0; j < 64; ++j) Ps[m][j] = (bf16_t)0.0f;
      } else {
        int ri = m / WIN, ci = m % WIN;
        int regI = 3 * region224(whh * WIN + ri) + region224(www * WIN + ci);
        float mx = -1e30f;
        for (int j = 0; j < NTOK; ++j) {
          int rj = j / WIN, cj = j % WIN;
          int regJ = 3 * region224(whh * WIN + rj) + region224(www * WIN + cj);
          float bias = rpb[((ri - rj + WIN - 1) * (2 * WIN - 1) + (ci - cj + WIN - 1)) * HEADS + h];
          float t = Ss[m][j] + bias + ((regJ != regI) ? -1e9f : 0.0f);
          Ss[m][j] = t;
          mx = fmaxf(mx, t);
        }
        float sum = 0.f;
        for (int j = 0; j < NTOK; ++j) { float e = expf(Ss[m][j] - mx); Ss[m][j] = e; sum += e; }
        float inv = 1.0f / sum;
        for (int j = 0; j < NTOK; ++j) Ps[m][j] = (bf16_t)(Ss[m][j] * inv);
        for (int j = NTOK; j < 64; ++j) Ps[m][j] = (bf16_t)0.0f;
      }
    }
  }
  __syncthreads();

  // O = P(64x64) * V(64x32): 4x2 tiles, two K-steps of 32
  v8f oacc[4][2] = {{{0.f}}};
  for (int ks = 0; ks < 64; ks += 32) {
    v16bf ap[4], bv[2];
#pragma unroll
    for (int tm = 0; tm < 4; ++tm) {
      int r = tm * 16 + fr;
      v8bf lo = *(const v8bf*)(&Ps[r][ks + half * 8]);
      v8bf hi = *(const v8bf*)(&Ps[r][ks + 16 + half * 8]);
      ap[tm] = __builtin_shufflevector(lo, hi, 0,1,2,3,4,5,6,7,8,9,10,11,12,13,14,15);
    }
#pragma unroll
    for (int tn = 0; tn < 2; ++tn) {
      v8bf lo = *(const v8bf*)(&Vt[tn * 16 + fr][ks + half * 8]);
      v8bf hi = *(const v8bf*)(&Vt[tn * 16 + fr][ks + 16 + half * 8]);
      bv[tn] = __builtin_shufflevector(lo, hi, 0,1,2,3,4,5,6,7,8,9,10,11,12,13,14,15);
    }
#pragma unroll
    for (int tm = 0; tm < 4; ++tm)
#pragma unroll
      for (int tn = 0; tn < 2; ++tn)
        oacc[tm][tn] = WMMA_BF16(ap[tm], bv[tn], oacc[tm][tn]);
  }
  // store rows i<49 to (token, head*32+c) layout for proj GEMM
#pragma unroll
  for (int tm = 0; tm < 4; ++tm)
#pragma unroll
    for (int tn = 0; tn < 2; ++tn)
#pragma unroll
      for (int r = 0; r < 8; ++r) {
        int i = tm * 16 + half * 8 + r;
        if (i < NTOK) {
          int c = tn * 16 + fr;
          attn_out[((size_t)(w * NTOK + i) * HEADS + h) * CH + c] = (bf16_t)oacc[tm][tn][r];
        }
      }
}

// ---------------- host: launch pipeline ----------------
extern "C" void kernel_launch(void* const* d_in, const int* in_sizes, int n_in,
                              void* d_out, int out_size, void* d_ws, size_t ws_size,
                              hipStream_t stream) {
  (void)in_sizes; (void)n_in; (void)out_size; (void)ws_size;
  const float* x      = (const float*)d_in[0];
  const float* n1g    = (const float*)d_in[1];
  const float* n1b    = (const float*)d_in[2];
  const float* qkv_w  = (const float*)d_in[3];
  const float* qkv_b  = (const float*)d_in[4];
  const float* rpb    = (const float*)d_in[5];
  const float* proj_w = (const float*)d_in[6];
  const float* proj_b = (const float*)d_in[7];
  const float* n2g    = (const float*)d_in[8];
  const float* n2b    = (const float*)d_in[9];
  const float* fc1_w  = (const float*)d_in[10];
  const float* fc1_b  = (const float*)d_in[11];
  const float* fc2_w  = (const float*)d_in[12];
  const float* fc2_b  = (const float*)d_in[13];
  float* out = (float*)d_out;

  char* ws = (char*)d_ws;
  constexpr size_t XW_OFF  = 0;                                    // xw / xn2 (77MB bf16)
  constexpr size_t QKV_OFF = (size_t)TOKENS * C_DIM * 2;           // qkv (231MB) / hid (308MB)
  constexpr size_t AO_OFF  = QKV_OFF + (size_t)TOKENS * 768 * 2;   // attn_out (77MB)
  constexpr size_t X2_OFF  = AO_OFF + (size_t)TOKENS * C_DIM * 2;  // x2 fp32 (154MB)
  constexpr size_t WT_OFF  = X2_OFF + (size_t)TOKENS * C_DIM * 4;  // bf16 weights (~0.9MB)
  bf16_t* xw    = (bf16_t*)(ws + XW_OFF);
  bf16_t* xn2   = (bf16_t*)(ws + XW_OFF);
  bf16_t* qkv   = (bf16_t*)(ws + QKV_OFF);
  bf16_t* hid   = (bf16_t*)(ws + QKV_OFF);
  bf16_t* ao    = (bf16_t*)(ws + AO_OFF);
  float*  x2    = (float*)(ws + X2_OFF);
  bf16_t* qkvWT = (bf16_t*)(ws + WT_OFF);
  bf16_t* projWT = qkvWT + 576 * 192;
  bf16_t* fc1WT  = projWT + 192 * 192;
  bf16_t* fc2WT  = fc1WT + 768 * 192;

  // weight transposes to (N,K) bf16
  transpose_to_bf16<<<(192 * 576 + 255) / 256, 256, 0, stream>>>(qkv_w, qkvWT, 192, 576);
  transpose_to_bf16<<<(192 * 192 + 255) / 256, 256, 0, stream>>>(proj_w, projWT, 192, 192);
  transpose_to_bf16<<<(192 * 768 + 255) / 256, 256, 0, stream>>>(fc1_w, fc1WT, 192, 768);
  transpose_to_bf16<<<(768 * 192 + 255) / 256, 256, 0, stream>>>(fc2_w, fc2WT, 768, 192);

  // LN1 + roll + window partition
  layernorm_kernel<true><<<TOKENS / 8, 256, 0, stream>>>(x, n1g, n1b, xw);
  // QKV GEMM (M=200704, N=576, K=192)
  gemm_bf16_kernel<QkvEpi><<<dim3(TOKENS / 128, 576 / 64), 256, 0, stream>>>(
      xw, qkvWT, 192, QkvEpi{qkv_b, qkv});
  // attention (24576 window-heads, 4 waves/block, 144KB dynamic LDS)
  attn_kernel<<<(NWTOT * HEADS) / 4, 128, 4 * AT_PER_WAVE, stream>>>(qkv, rpb, ao);
  // proj GEMM + window reverse + roll back + residual
  gemm_bf16_kernel<ProjEpi><<<dim3(TOKENS / 128, 192 / 64), 256, 0, stream>>>(
      ao, projWT, 192, ProjEpi{proj_b, x, x2});
  // LN2
  layernorm_kernel<false><<<TOKENS / 8, 256, 0, stream>>>(x2, n2g, n2b, xn2);
  // FC1 + GELU (N=768)
  gemm_bf16_kernel<Fc1Epi><<<dim3(TOKENS / 128, 768 / 64), 256, 0, stream>>>(
      xn2, fc1WT, 192, Fc1Epi{fc1_b, hid});
  // FC2 + residual (K=768)
  gemm_bf16_kernel<Fc2Epi><<<dim3(TOKENS / 128, 192 / 64), 256, 0, stream>>>(
      hid, fc2WT, 768, Fc2Epi{fc2_b, x2, out});
}
